// Seq2SeqDualModelAlign_3264175145498
// MI455X (gfx1250) — matrix operand order repaired and checked by reading, not compiled, not run on hardware
//
#include <hip/hip_runtime.h>
#include <hip/hip_bf16.h>

// ---------------------------------------------------------------------------
// CDNA5 (gfx1250) bf16 WMMA implementation of the dual-attention seq2seq model
// wave32 everywhere; v_wmma_f32_16x16x32_bf16 for all GEMMs;
// GLOBAL_LOAD_ASYNC_TO_LDS_B128 (ASYNCcnt) for the align_prob LDS staging.
// ---------------------------------------------------------------------------

typedef __bf16 bf16x8  __attribute__((ext_vector_type(8)));
typedef __bf16 bf16x16 __attribute__((ext_vector_type(16)));
typedef float  f32x8   __attribute__((ext_vector_type(8)));

#define DEVI __device__ __forceinline__

// model dims
constexpr int cV = 20000, cSE = 400, cFR = 60, cHH = 250, cH = 500;
constexpr int cB = 32, cTE = 100, cTD = 60;
// padded dims (K multiples of 32, N multiples of 16)
constexpr int pKE = 480;           // enc input 460 -> 480
constexpr int pGE = 1008;          // enc gates 1000 -> 1008
constexpr int pKHH = 256;          // HH 250 -> 256
constexpr int pKD = 416;           // dec input 400 -> 416
constexpr int pGD = 2000;          // dec gates (125 tiles exactly)
constexpr int pKH = 512;           // H 500 -> 512
constexpr int pKT = 128;           // TE 100 -> 128
constexpr int pTE = 112;           // TE as N: 100 -> 112
constexpr int pTD = 64;            // TD as M: 60 -> 64
constexpr int pK2H = 1024;         // 2H 1000 -> 1024
constexpr int NCH = 80;            // vocab chunk (5 N-tiles)
constexpr int NCHN = cV / NCH;     // 250 chunks

// d_out layout (floats)
constexpr size_t OFF_HT = 38400000ull;
constexpr size_t OFF_CT = 38416000ull;
constexpr size_t OFF_AP = 38432000ull;
constexpr size_t OFF_DP = 39200000ull;

// ---------------------------------------------------------------------------
DEVI unsigned short f2bf(float f) {
    unsigned int u = __float_as_uint(f);
    u += 0x7FFFu + ((u >> 16) & 1u);        // round-to-nearest-even
    return (unsigned short)(u >> 16);
}
DEVI float sigmf(float x) { return 1.f / (1.f + expf(-x)); }

union BfV { bf16x16 v; bf16x8 h[2]; };

// A-matrix 16x32 bf16 tile per ISA layout: lanes<16 hold K {0..7,16..23},
// lanes>=16 hold K {8..15,24..31}; row M = lane&15. A row-major [M][lda].
DEVI bf16x16 load_A_tile(const unsigned short* A, int lda, int lane) {
    const int m  = lane & 15;
    const int ko = (lane < 16) ? 0 : 8;
    const unsigned short* p = A + m * lda + ko;
    BfV u;
    u.h[0] = *(const bf16x8*)(p);
    u.h[1] = *(const bf16x8*)(p + 16);
    return u.v;
}
// B-matrix 32x16 bf16 tile: lanes<16 hold K {0..7,16..23}, lanes>=16 hold K {8..15,24..31},
// column N = lane&15. B stored TRANSPOSED row-major: Bt[N][ldb].
DEVI bf16x16 load_B_tile(const unsigned short* Bt, int ldb, int lane) {
    const int n  = lane & 15;
    const int ko = (lane < 16) ? 0 : 8;
    const unsigned short* p = Bt + n * ldb + ko;
    BfV u;
    u.h[0] = *(const bf16x8*)(p);
    u.h[1] = *(const bf16x8*)(p + 16);
    return u.v;
}
DEVI f32x8 wmma_bf16(bf16x16 a, bf16x16 b, f32x8 c) {
    return __builtin_amdgcn_wmma_f32_16x16x32_bf16(false, a, false, b,
                                                   (short)0, c, false, false);
}
// one 16x16 C tile: accumulate A[m0..m0+15][0..K) * Bt[n0..n0+15][0..K)^T
DEVI f32x8 gemm_accum(const unsigned short* A, int lda,
                      const unsigned short* Bt, int ldb,
                      int K, f32x8 acc, int lane) {
    for (int k = 0; k < K; k += 32) {
        bf16x16 a = load_A_tile(A + k, lda, lane);
        bf16x16 b = load_B_tile(Bt + k, ldb, lane);
        acc = wmma_bf16(a, b, acc);
    }
    return acc;
}
DEVI f32x8 v8zero() { f32x8 z = {0.f,0.f,0.f,0.f,0.f,0.f,0.f,0.f}; return z; }

DEVI float wred_max(float v) { for (int o = 16; o; o >>= 1) v = fmaxf(v, __shfl_xor(v, o, 32)); return v; }
DEVI float wred_sum(float v) { for (int o = 16; o; o >>= 1) v += __shfl_xor(v, o, 32); return v; }

// ---- CDNA5 async global->LDS path (ASYNCcnt) -------------------------------
// raw wave-relative LDS byte address of a __shared__ object
DEVI unsigned lds_byte_addr(const void* p) {
    return (unsigned)(unsigned long long)
        (__attribute__((address_space(3))) const void*)p;
}
// GLOBAL_LOAD_ASYNC_TO_LDS_B128: one 16B granule per lane, tracked by ASYNCcnt
DEVI void async_g2l_b128(unsigned lds_addr, const void* gaddr) {
    asm volatile("global_load_async_to_lds_b128 %0, %1, off"
                 :: "v"(lds_addr), "v"(gaddr) : "memory");
}
DEVI void wait_asynccnt0() {
    asm volatile("s_wait_asynccnt 0" ::: "memory");
}

// ---------------------------------------------------------------------------
__global__ void k_zero(unsigned int* p, long long n) {
    long long stride = (long long)gridDim.x * blockDim.x;
    for (long long i = (long long)blockIdx.x * blockDim.x + threadIdx.x; i < n; i += stride)
        p[i] = 0u;
}

// f32 [R][K] -> bf16 [R][ldd] (pads pre-zeroed)
__global__ void k_conv(const float* __restrict__ src, unsigned short* __restrict__ dst,
                       int R, int K, int ldd) {
    long long n = (long long)R * K, stride = (long long)gridDim.x * blockDim.x;
    for (long long i = (long long)blockIdx.x * blockDim.x + threadIdx.x; i < n; i += stride) {
        int r = (int)(i / K), c = (int)(i % K);
        dst[(size_t)r * ldd + c] = f2bf(src[i]);
    }
}
// f32 [K][N] -> bf16 transposed [N][ldd]
__global__ void k_convT(const float* __restrict__ src, unsigned short* __restrict__ dst,
                        int K, int N, int ldd) {
    long long n = (long long)K * N, stride = (long long)gridDim.x * blockDim.x;
    for (long long i = (long long)blockIdx.x * blockDim.x + threadIdx.x; i < n; i += stride) {
        int k = (int)(i / N), c = (int)(i % N);
        dst[(size_t)c * ldd + k] = f2bf(src[i]);
    }
}

// encoder input: enc_in bf16 [TE*B][480] (row = t*B+b) + input_z A [B*112][64]
__global__ void k_enc_in(const int* __restrict__ value, const int* __restrict__ field,
                         const int* __restrict__ ppos, const int* __restrict__ pneg,
                         const float* __restrict__ es, const float* __restrict__ ef,
                         const float* __restrict__ ep, const float* __restrict__ en,
                         unsigned short* __restrict__ encin, unsigned short* __restrict__ inzA) {
    long long n = (long long)cTE * cB * 460, stride = (long long)gridDim.x * blockDim.x;
    for (long long i = (long long)blockIdx.x * blockDim.x + threadIdx.x; i < n; i += stride) {
        int row = (int)(i / 460), c = (int)(i % 460);
        int t = row / cB, b = row % cB;
        float v;
        if (c < 400)      v = es[(size_t)value[b * cTE + t] * cSE + c];
        else if (c < 450) v = ef[(size_t)field[b * cTE + t] * 50 + (c - 400)];
        else if (c < 455) v = ep[(size_t)ppos[b * cTE + t] * 5 + (c - 450)];
        else              v = en[(size_t)pneg[b * cTE + t] * 5 + (c - 455)];
        unsigned short bf = f2bf(v);
        encin[(size_t)row * pKE + c] = bf;
        if (c >= 400) inzA[((size_t)b * pTE + t) * 64 + (c - 400)] = bf;
    }
}

// decoder input embeddings: bf16 [TD*B][416] (row = d*B+b)
__global__ void k_sent_e(const int* __restrict__ sent, const float* __restrict__ es,
                         unsigned short* __restrict__ dst) {
    long long n = (long long)cTD * cB * cSE, stride = (long long)gridDim.x * blockDim.x;
    for (long long i = (long long)blockIdx.x * blockDim.x + threadIdx.x; i < n; i += stride) {
        int row = (int)(i / cSE), c = (int)(i % cSE);
        dst[(size_t)row * pKD + c] = f2bf(es[(size_t)sent[(row % cB) * cTD + (row / cB)] * cSE + c]);
    }
}

// generic tiled bf16 GEMM: C[M,N] = A[M,Kp] x Bt[N,Kp]^T (+bias). one wave/tile.
__global__ __launch_bounds__(256)
void k_gemm(const unsigned short* __restrict__ A, int lda,
            const unsigned short* __restrict__ Bt, int ldb,
            int M, int N, int K,
            float* __restrict__ Cf, int ldc,
            unsigned short* __restrict__ Cb, int ldcb,
            const float* __restrict__ bias) {
    const int lane = threadIdx.x & 31;
    const int wid  = blockIdx.x * (blockDim.x >> 5) + (threadIdx.x >> 5);
    const int nw   = gridDim.x * (blockDim.x >> 5);
    const int tm = M >> 4, tn = (N + 15) >> 4;
    for (int t = wid; t < tm * tn; t += nw) {
        const int ti = t / tn, tj = t % tn;
        const int m0 = ti << 4, n0 = tj << 4;
        f32x8 acc = v8zero();
        acc = gemm_accum(A + (size_t)m0 * lda, lda, Bt + (size_t)n0 * ldb, ldb, K, acc, lane);
        const int n = n0 + (lane & 15);
        if (n < N) {
            const float bv = bias ? bias[n] : 0.f;
            #pragma unroll
            for (int j = 0; j < 8; ++j) {
                int m = m0 + j + ((lane >> 4) << 3);
                float v = acc[j] + bv;
                if (Cf) Cf[(size_t)m * ldc + n] = v;
                if (Cb) Cb[(size_t)m * ldcb + n] = f2bf(v);
            }
        }
    }
}

// sequential BiLSTM encoder: 1 WG per direction, 16 waves, WMMA for h@Whh^T
__global__ __launch_bounds__(512)
void k_enc_rec(const float* __restrict__ gxF, const float* __restrict__ gxB,
               const unsigned short* __restrict__ WhhF, const unsigned short* __restrict__ WhhB,
               float* __restrict__ hst, float* __restrict__ cst, unsigned short* __restrict__ hbst,
               float* __restrict__ gst,
               unsigned short* __restrict__ encA, unsigned short* __restrict__ encT,
               float* __restrict__ hfin, float* __restrict__ cfin) {
    const int dir = blockIdx.x;
    const float* gx = dir ? gxB : gxF;
    const unsigned short* Whh = dir ? WhhB : WhhF;
    float* h = hst + dir * cB * cHH;
    float* c = cst + dir * cB * cHH;
    unsigned short* hB = hbst + dir * cB * pKHH;
    float* g = gst + dir * cB * pGE;
    const int tid = threadIdx.x, lane = tid & 31, wid = tid >> 5;

    for (int i = tid; i < cB * cHH; i += 512) { h[i] = 0.f; c[i] = 0.f; }
    for (int i = tid; i < cB * pKHH; i += 512) hB[i] = 0;
    __threadfence(); __syncthreads();

    for (int s = 0; s < cTE; ++s) {
        const int t_in = dir ? (cTE - 1 - s) : s;
        const float* gxt = gx + (size_t)t_in * cB * pGE;
        for (int t = wid; t < 2 * 63; t += 16) {        // 2 M-tiles x 63 N-tiles
            const int ti = t / 63, tj = t % 63;
            const int m0 = ti << 4, n0 = tj << 4;
            f32x8 acc;
            const int n = n0 + (lane & 15);
            #pragma unroll
            for (int j = 0; j < 8; ++j)
                acc[j] = gxt[(size_t)(m0 + j + ((lane >> 4) << 3)) * pGE + n];
            acc = gemm_accum(hB + m0 * pKHH, pKHH, Whh + (size_t)n0 * pKHH, pKHH, pKHH, acc, lane);
            #pragma unroll
            for (int j = 0; j < 8; ++j)
                g[(size_t)(m0 + j + ((lane >> 4) << 3)) * pGE + n] = acc[j];
        }
        __threadfence(); __syncthreads();
        for (int i = tid; i < cB * cHH; i += 512) {     // gate order i,f,g,o
            const int b = i / cHH, j = i % cHH;
            const float* gr = g + (size_t)b * pGE;
            float cn = sigmf(gr[cHH + j]) * c[i] + sigmf(gr[j]) * tanhf(gr[2 * cHH + j]);
            float hn = sigmf(gr[3 * cHH + j]) * tanhf(cn);
            c[i] = cn; h[i] = hn;
            unsigned short hb = f2bf(hn);
            hB[b * pKHH + j] = hb;
            encA[((size_t)b * pTE + t_in) * pKH + dir * cHH + j] = hb;
            encT[((size_t)b * pKH + dir * cHH + j) * pKT + t_in] = hb;
        }
        __threadfence(); __syncthreads();
    }
    for (int i = tid; i < cB * cHH; i += 512) {
        hfin[dir * cB * cHH + i] = h[i];
        cfin[dir * cB * cHH + i] = c[i];
    }
}

// torch .view(1,B,2*HH) quirk on [2,B,HH] stacked finals -> dh0/dc0
__global__ void k_dh0(const float* __restrict__ hfin, const float* __restrict__ cfin,
                      float* __restrict__ h, float* __restrict__ c, unsigned short* __restrict__ hB) {
    long long n = (long long)cB * cH, stride = (long long)gridDim.x * blockDim.x;
    for (long long i = (long long)blockIdx.x * blockDim.x + threadIdx.x; i < n; i += stride) {
        int b = (int)(i / cH), off = (int)(i % cH);
        int half = off >= cHH, k = off - half * cHH;
        int src = (b < 16) ? (2 * b + half) * cHH + k
                           : cB * cHH + (2 * (b - 16) + half) * cHH + k;
        float hv = hfin[src], cv = cfin[src];
        h[i] = hv; c[i] = cv;
        hB[b * pKH + off] = f2bf(hv);
    }
}

// sequential decoder LSTM: 1 WG, 16 waves, WMMA h@Whh^T
__global__ __launch_bounds__(512)
void k_dec_rec(const float* __restrict__ gxd, const unsigned short* __restrict__ Whh,
               float* __restrict__ h, float* __restrict__ c, unsigned short* __restrict__ hB,
               float* __restrict__ g, unsigned short* __restrict__ hsA,
               unsigned short* __restrict__ catA, float* __restrict__ out) {
    const int tid = threadIdx.x, lane = tid & 31, wid = tid >> 5;
    for (int d = 0; d < cTD; ++d) {
        const float* gxt = gxd + (size_t)d * cB * pGD;
        for (int t = wid; t < 2 * 125; t += 16) {
            const int ti = t / 125, tj = t % 125;
            const int m0 = ti << 4, n0 = tj << 4;
            const int n = n0 + (lane & 15);
            f32x8 acc;
            #pragma unroll
            for (int j = 0; j < 8; ++j)
                acc[j] = gxt[(size_t)(m0 + j + ((lane >> 4) << 3)) * pGD + n];
            acc = gemm_accum(hB + m0 * pKH, pKH, Whh + (size_t)n0 * pKH, pKH, pKH, acc, lane);
            #pragma unroll
            for (int j = 0; j < 8; ++j)
                g[(size_t)(m0 + j + ((lane >> 4) << 3)) * pGD + n] = acc[j];
        }
        __threadfence(); __syncthreads();
        for (int i = tid; i < cB * cH; i += 512) {
            const int b = i / cH, j = i % cH;
            const float* gr = g + (size_t)b * pGD;
            float cn = sigmf(gr[cH + j]) * c[i] + sigmf(gr[j]) * tanhf(gr[2 * cH + j]);
            float hn = sigmf(gr[3 * cH + j]) * tanhf(cn);
            c[i] = cn; h[i] = hn;
            unsigned short hb = f2bf(hn);
            hB[b * pKH + j] = hb;
            hsA[((size_t)b * pTD + d) * pKH + j] = hb;
            catA[((size_t)b * pTD + d) * pK2H + j] = hb;   // [h | ctx] left half
        }
        __threadfence(); __syncthreads();
    }
    for (int i = tid; i < cB * cH; i += 512) {
        out[OFF_HT + i] = h[i];
        out[OFF_CT + i] = c[i];
    }
}

// per-batch dual attention + context + output projection
__global__ __launch_bounds__(512)
void k_att(const unsigned short* __restrict__ hsA, const unsigned short* __restrict__ Ea,
           const unsigned short* __restrict__ Ez, const unsigned short* __restrict__ encT,
           unsigned short* __restrict__ attnA, unsigned short* __restrict__ catA,
           const unsigned short* __restrict__ WoT, unsigned short* __restrict__ decA,
           const int* __restrict__ value, const float* __restrict__ emb_target,
           float* __restrict__ out) {
    __shared__ __align__(16) float sSA[pTD * pTE];
    __shared__ __align__(16) float sSZ[pTD * pTE];
    __shared__ int sWidx[pTD];
    const int b = blockIdx.x;
    const int tid = threadIdx.x, lane = tid & 31, wid = tid >> 5;
    const unsigned short* Ah   = hsA  + (size_t)b * pTD * pKH;
    const unsigned short* BtEa = Ea   + (size_t)b * pTE * pKH;
    const unsigned short* BtEz = Ez   + (size_t)b * pTE * pKH;

    // sa and sz scores: [64,112] each, K=512; 2*28 tiles
    for (int t = wid; t < 56; t += 16) {
        const int which = t >= 28, tt = t - which * 28;
        const int ti = tt / 7, tj = tt % 7;
        const unsigned short* Bt = which ? BtEz : BtEa;
        f32x8 acc = v8zero();
        acc = gemm_accum(Ah + ti * 16 * pKH, pKH, Bt + (size_t)tj * 16 * pKH, pKH, pKH, acc, lane);
        float* dst = which ? sSZ : sSA;
        const int n = tj * 16 + (lane & 15);
        #pragma unroll
        for (int j = 0; j < 8; ++j)
            dst[(ti * 16 + j + ((lane >> 4) << 3)) * pTE + n] = acc[j];
    }
    __syncthreads();

    // dual softmax + renorm + argmax, one wave per decoder row
    for (int d = wid; d < cTD; d += 16) {
        float* ra = sSA + d * pTE;
        float* rz = sSZ + d * pTE;
        float ma = -1e30f, mz = -1e30f;
        for (int t = lane; t < cTE; t += 32) { ma = fmaxf(ma, ra[t]); mz = fmaxf(mz, rz[t]); }
        ma = wred_max(ma); mz = wred_max(mz);
        float sa = 0.f, sz = 0.f;
        for (int t = lane; t < cTE; t += 32) { sa += expf(ra[t] - ma); sz += expf(rz[t] - mz); }
        sa = wred_sum(sa); sz = wred_sum(sz);
        float asum = 0.f;
        for (int t = lane; t < cTE; t += 32) {
            float a = (expf(ra[t] - ma) / sa) * (expf(rz[t] - mz) / sz);
            ra[t] = a; asum += a;
        }
        asum = wred_sum(asum);
        const float inv = 1.f / (asum + 1e-8f);
        float bv = -1.f; int bi = 0;
        for (int t = lane; t < cTE; t += 32) {
            float a = ra[t] * inv;
            ra[t] = a;
            attnA[((size_t)b * pTD + d) * pKT + t] = f2bf(a);
            if (a > bv) { bv = a; bi = t; }
        }
        for (int o = 16; o; o >>= 1) {
            float ov = __shfl_xor(bv, o, 32);
            int   oi = __shfl_xor(bi, o, 32);
            if (ov > bv || (ov == bv && oi < bi)) { bv = ov; bi = oi; }
        }
        if (lane == 0) sWidx[d] = value[b * cTE + bi];
    }
    __syncthreads();

    // attn_pred gather
    for (int i = tid; i < cTD * cSE; i += 512) {
        const int d = i / cSE, k = i % cSE;
        out[OFF_AP + ((size_t)b * cTD + d) * cSE + k] = emb_target[(size_t)sWidx[d] * cSE + k];
    }
    __threadfence(); __syncthreads();

    // ctx = attn @ enc_out : A=attnA[b] [64,128], Bt=encT[b] [512,128]
    const unsigned short* Aat = attnA + (size_t)b * pTD * pKT;
    const unsigned short* BtE = encT  + (size_t)b * pKH * pKT;
    for (int t = wid; t < 4 * 32; t += 16) {
        const int ti = t >> 5, tj = t & 31;
        f32x8 acc = v8zero();
        acc = gemm_accum(Aat + ti * 16 * pKT, pKT, BtE + (size_t)tj * 16 * pKT, pKT, pKT, acc, lane);
        const int n = tj * 16 + (lane & 15);
        if (n < cH)
            #pragma unroll
            for (int j = 0; j < 8; ++j)
                catA[((size_t)b * pTD + ti * 16 + j + ((lane >> 4) << 3)) * pK2H + cH + n] = f2bf(acc[j]);
    }
    __threadfence(); __syncthreads();

    // o = tanh([h|ctx] @ Wo) : A=catA[b] [64,1024], Bt=WoT [512,1024]
    const unsigned short* Ac = catA + (size_t)b * pTD * pK2H;
    for (int t = wid; t < 4 * 32; t += 16) {
        const int ti = t >> 5, tj = t & 31;
        f32x8 acc = v8zero();
        acc = gemm_accum(Ac + ti * 16 * pK2H, pK2H, WoT + (size_t)tj * 16 * pK2H, pK2H, pK2H, acc, lane);
        const int n = tj * 16 + (lane & 15);
        if (n < cH)
            #pragma unroll
            for (int j = 0; j < 8; ++j)
                decA[((size_t)b * pTD + ti * 16 + j + ((lane >> 4) << 3)) * pKH + n] = f2bf(tanhf(acc[j]));
    }
}

// fused logits + p_lex over an 80-wide vocab chunk.
// align_prob tile staged global->LDS with GLOBAL_LOAD_ASYNC_TO_LDS_B128,
// then transposed/converted to bf16 in LDS for the WMMA B operand.
__global__ __launch_bounds__(256)
void k_big(const unsigned short* __restrict__ decA, const unsigned short* __restrict__ linWT,
           const float* __restrict__ linb, const unsigned short* __restrict__ attnA,
           const float* __restrict__ align_prob, const float* __restrict__ xpar,
           float* __restrict__ logits, float* __restrict__ uout) {
    __shared__ __align__(16) float sF[cTE * NCH];              // 32000 B f32 staging
    __shared__ __align__(16) unsigned short sB[NCH * pKT];     // 20480 B bf16 B-tile
    const int b = blockIdx.x / NCHN, ch = blockIdx.x % NCHN, v0 = ch * NCH;
    const int tid = threadIdx.x, lane = tid & 31, wid = tid >> 5;

    // async stage: 100 rows x 80 f32 = 2000 granules of 16B, global -> LDS
    const float* gbase = align_prob + (size_t)b * cTE * cV + v0;
    for (int gi = tid; gi < cTE * (NCH / 4); gi += 256) {
        const int k = gi / (NCH / 4), j = (gi % (NCH / 4)) * 4;
        async_g2l_b128(lds_byte_addr(&sF[k * NCH + j]), gbase + (size_t)k * cV + j);
    }
    // streaming prefetch of the next chunk's rows
    if (v0 + NCH < cV && tid < cTE)
        __builtin_prefetch(align_prob + ((size_t)b * cTE + tid) * cV + v0 + NCH, 0, 1);
    wait_asynccnt0();
    __syncthreads();
    // transpose + convert to bf16 Bt layout [n][k] (k >= 100 zero-padded)
    for (int i = tid; i < NCH * pKT; i += 256) {
        const int n = i >> 7, k = i & 127;
        sB[i] = f2bf(k < cTE ? sF[k * NCH + n] : 0.f);
    }
    __syncthreads();

    const unsigned short* Ad = decA  + (size_t)b * pTD * pKH;
    const unsigned short* Aa = attnA + (size_t)b * pTD * pKT;
    for (int t = wid; t < 4 * 5; t += 8) {
        const int ti = t / 5, tj = t % 5;
        f32x8 aL = v8zero();
        aL = gemm_accum(Ad + ti * 16 * pKH, pKH, linWT + (size_t)(v0 + tj * 16) * pKH, pKH, pKH, aL, lane);
        f32x8 aP = v8zero();
        aP = gemm_accum(Aa + ti * 16 * pKT, pKT, (const unsigned short*)sB + tj * 16 * pKT, pKT, pKT, aP, lane);
        const int v = v0 + tj * 16 + (lane & 15);
        const float lb = linb[v];
        const float lam = sigmf(xpar[v]);
        #pragma unroll
        for (int j = 0; j < 8; ++j) {
            const int d = ti * 16 + j + ((lane >> 4) << 3);
            if (d < cTD) {
                const size_t idx = ((size_t)b * cTD + d) * cV + v;
                const float lg = aL[j] + lb;
                logits[idx] = lg;
                uout[idx] = lam * aP[j] + (1.f - lam) * lg;   // u = lam*p_lex + (1-lam)*logit
            }
        }
    }
}

// per-row log-sum-exp of logits
__global__ __launch_bounds__(256)
void k_lse(const float* __restrict__ logits, float* __restrict__ logZ) {
    const int r = blockIdx.x, tid = threadIdx.x;
    const float* row = logits + (size_t)r * cV;
    __shared__ float red[256];
    float m = -1e30f;
    for (int i = tid; i < cV; i += 256) m = fmaxf(m, row[i]);
    red[tid] = m; __syncthreads();
    for (int s = 128; s; s >>= 1) { if (tid < s) red[tid] = fmaxf(red[tid], red[tid + s]); __syncthreads(); }
    m = red[0]; __syncthreads();
    float sum = 0.f;
    for (int i = tid; i < cV; i += 256) sum += expf(row[i] - m);
    red[tid] = sum; __syncthreads();
    for (int s = 128; s; s >>= 1) { if (tid < s) red[tid] += red[tid + s]; __syncthreads(); }
    if (tid == 0) logZ[r] = m + logf(red[0]);
}

// final log-softmax of z = u - (1-lam)*logZ1, in-place in d_out, + decoder_pred
__global__ __launch_bounds__(256)
void k_final(float* __restrict__ u, const float* __restrict__ logZ1,
             const float* __restrict__ xpar, const float* __restrict__ emb_target,
             float* __restrict__ out) {
    const int r = blockIdx.x, tid = threadIdx.x;
    float* row = u + (size_t)r * cV;
    const float lz1 = logZ1[r];
    __shared__ float sm[256], ss[256], sbm[256];
    __shared__ int sbi[256];
    float m = -1e30f, s = 0.f, bm = -1e30f;
    int bi = 0;
    for (int i = tid; i < cV; i += 256) {
        const float lam = sigmf(xpar[i]);
        const float z = row[i] - (1.f - lam) * lz1;
        if (z > m) { s = s * expf(m - z) + 1.f; m = z; } else s += expf(z - m);
        if (z > bm || (z == bm && i < bi)) { bm = z; bi = i; }
    }
    sm[tid] = m; ss[tid] = s; sbm[tid] = bm; sbi[tid] = bi;
    __syncthreads();
    for (int st = 128; st; st >>= 1) {
        if (tid < st) {
            const float m2 = sm[tid + st], s2 = ss[tid + st];
            if (m2 > sm[tid]) { ss[tid] = ss[tid] * expf(sm[tid] - m2) + s2; sm[tid] = m2; }
            else              { ss[tid] += s2 * expf(m2 - sm[tid]); }
            const float b2 = sbm[tid + st]; const int i2 = sbi[tid + st];
            if (b2 > sbm[tid] || (b2 == sbm[tid] && i2 < sbi[tid])) { sbm[tid] = b2; sbi[tid] = i2; }
        }
        __syncthreads();
    }
    const float logZ2 = sm[0] + logf(ss[0]);
    const int amax = sbi[0];
    for (int i = tid; i < cV; i += 256) {
        const float lam = sigmf(xpar[i]);
        row[i] = (row[i] - (1.f - lam) * lz1) - logZ2;     // p_bias
    }
    for (int i = tid; i < cSE; i += 256)
        out[OFF_DP + (size_t)r * cSE + i] = emb_target[(size_t)amax * cSE + i];
}

// ---------------------------------------------------------------------------
extern "C" void kernel_launch(void* const* d_in, const int* in_sizes, int n_in,
                              void* d_out, int out_size, void* d_ws, size_t ws_size,
                              hipStream_t stream) {
    (void)in_sizes; (void)n_in; (void)out_size; (void)ws_size;
    const int*   sent   = (const int*)d_in[0];
    const int*   value  = (const int*)d_in[1];
    const int*   field  = (const int*)d_in[2];
    const int*   ppos   = (const int*)d_in[3];
    const int*   pneg   = (const int*)d_in[4];
    const float* aprob  = (const float*)d_in[7];
    const float* e_sent = (const float*)d_in[8];
    const float* e_fld  = (const float*)d_in[9];
    const float* e_pp   = (const float*)d_in[10];
    const float* e_pn   = (const float*)d_in[11];
    const float* e_tgt  = (const float*)d_in[12];
    const float* WihF   = (const float*)d_in[13];
    const float* WhhF   = (const float*)d_in[14];
    const float* bF     = (const float*)d_in[15];
    const float* WihB   = (const float*)d_in[16];
    const float* WhhB   = (const float*)d_in[17];
    const float* bB     = (const float*)d_in[18];
    const float* dWih   = (const float*)d_in[19];
    const float* dWhh   = (const float*)d_in[20];
    const float* db     = (const float*)d_in[21];
    const float* Wa     = (const float*)d_in[22];
    const float* Wz     = (const float*)d_in[23];
    const float* Wo     = (const float*)d_in[24];
    const float* linW   = (const float*)d_in[25];
    const float* linb   = (const float*)d_in[26];
    const float* xpar   = (const float*)d_in[27];
    float* out = (float*)d_out;

    // ---- workspace layout (256B aligned) ----
    char* w = (char*)d_ws;
    size_t off = 0;
    auto take = [&](size_t elems, size_t esz) {
        size_t o = off; off += ((elems * esz + 255) & ~(size_t)255); return o;
    };
    const size_t oEncIn  = take((size_t)cTE * cB * pKE, 2);
    const size_t oWihF   = take((size_t)pGE * pKE, 2);
    const size_t oWihB   = take((size_t)pGE * pKE, 2);
    const size_t oWhhF   = take((size_t)pGE * pKHH, 2);
    const size_t oWhhB   = take((size_t)pGE * pKHH, 2);
    const size_t oSentE  = take((size_t)cTD * cB * pKD, 2);
    const size_t oDWih   = take((size_t)pGD * pKD, 2);
    const size_t oDWhh   = take((size_t)pGD * pKH, 2);
    const size_t oInzA   = take((size_t)cB * pTE * 64, 2);
    const size_t oWaT    = take((size_t)pKH * pKH, 2);
    const size_t oWzT    = take((size_t)pKH * 64, 2);
    const size_t oWoT    = take((size_t)pKH * pK2H, 2);
    const size_t oLinWT  = take((size_t)cV * pKH, 2);
    const size_t oEncA   = take((size_t)cB * pTE * pKH, 2);
    const size_t oEncT   = take((size_t)cB * pKH * pKT, 2);
    const size_t oEA     = take((size_t)cB * pTE * pKH, 2);
    const size_t oEZ     = take((size_t)cB * pTE * pKH, 2);
    const size_t oHsA    = take((size_t)cB * pTD * pKH, 2);
    const size_t oCatA   = take((size_t)cB * pTD * pK2H, 2);
    const size_t oAttnA  = take((size_t)cB * pTD * pKT, 2);
    const size_t oDecA   = take((size_t)cB * pTD * pKH, 2);
    const size_t oEH     = take((size_t)2 * cB * cHH, 4);
    const size_t oEC     = take((size_t)2 * cB * cHH, 4);
    const size_t oEHB    = take((size_t)2 * cB * pKHH, 2);
    const size_t oEG     = take((size_t)2 * cB * pGE, 4);
    const size_t oHfin   = take((size_t)2 * cB * cHH, 4);
    const size_t oCfin   = take((size_t)2 * cB * cHH, 4);
    const size_t oDH     = take((size_t)cB * cH, 4);
    const size_t oDC     = take((size_t)cB * cH, 4);
    const size_t oDHB    = take((size_t)cB * pKH, 2);
    const size_t oDG     = take((size_t)cB * pGD, 4);
    const size_t oLogZ   = take((size_t)cB * cTD, 4);
    const size_t oGxF    = take((size_t)cTE * cB * pGE, 4);
    const size_t oGxB    = take((size_t)cTE * cB * pGE, 4);
    const size_t oGxD    = take((size_t)cTD * cB * pGD, 4);
    const size_t zBytes  = off;                    // everything above needs zero pads
    const size_t oLogits = take((size_t)cB * cTD * cV, 4);

    #define WSU(o) ((unsigned short*)(w + (o)))
    #define WSF(o) ((float*)(w + (o)))

    // 1) zero all pad-bearing scratch
    k_zero<<<4096, 256, 0, stream>>>((unsigned int*)d_ws, (long long)(zBytes / 4));

    // 2) bf16 weight conversions (Bt = weight stored as [N][Kpad])
    auto cgrid = [](long long n) { long long g = (n + 255) / 256; return (int)(g > 8192 ? 8192 : g); };
    k_conv <<<cgrid(1000LL*460), 256, 0, stream>>>(WihF, WSU(oWihF), 1000, 460, pKE);
    k_conv <<<cgrid(1000LL*460), 256, 0, stream>>>(WihB, WSU(oWihB), 1000, 460, pKE);
    k_conv <<<cgrid(1000LL*250), 256, 0, stream>>>(WhhF, WSU(oWhhF), 1000, 250, pKHH);
    k_conv <<<cgrid(1000LL*250), 256, 0, stream>>>(WhhB, WSU(oWhhB), 1000, 250, pKHH);
    k_conv <<<cgrid(2000LL*400), 256, 0, stream>>>(dWih, WSU(oDWih), 2000, 400, pKD);
    k_conv <<<cgrid(2000LL*500), 256, 0, stream>>>(dWhh, WSU(oDWhh), 2000, 500, pKH);
    k_convT<<<cgrid(500LL*500),  256, 0, stream>>>(Wa,   WSU(oWaT),  500, 500, pKH);
    k_convT<<<cgrid(60LL*500),   256, 0, stream>>>(Wz,   WSU(oWzT),  60,  500, 64);
    k_convT<<<cgrid(1000LL*500), 256, 0, stream>>>(Wo,   WSU(oWoT),  1000, 500, pK2H);
    k_convT<<<cgrid(500LL*20000),256, 0, stream>>>(linW, WSU(oLinWT),500, cV, pKH);

    // 3) embedding gathers
    k_enc_in<<<cgrid((long long)cTE*cB*460), 256, 0, stream>>>(
        value, field, ppos, pneg, e_sent, e_fld, e_pp, e_pn, WSU(oEncIn), WSU(oInzA));
    k_sent_e<<<cgrid((long long)cTD*cB*cSE), 256, 0, stream>>>(sent, e_sent, WSU(oSentE));

    // 4) input-side gate GEMMs (hoisted out of the recurrences)
    auto ggrid = [](int tiles) { return (tiles + 7) / 8; };
    k_gemm<<<ggrid(200*63), 256, 0, stream>>>(WSU(oEncIn), pKE, WSU(oWihF), pKE,
        cTE*cB, 1000, pKE, WSF(oGxF), pGE, (unsigned short*)nullptr, 0, bF);
    k_gemm<<<ggrid(200*63), 256, 0, stream>>>(WSU(oEncIn), pKE, WSU(oWihB), pKE,
        cTE*cB, 1000, pKE, WSF(oGxB), pGE, (unsigned short*)nullptr, 0, bB);
    k_gemm<<<ggrid(120*125), 256, 0, stream>>>(WSU(oSentE), pKD, WSU(oDWih), pKD,
        cTD*cB, 2000, pKD, WSF(oGxD), pGD, (unsigned short*)nullptr, 0, db);

    // 5) BiLSTM encoder (2 WGs, one per direction)
    k_enc_rec<<<2, 512, 0, stream>>>(WSF(oGxF), WSF(oGxB), WSU(oWhhF), WSU(oWhhB),
        WSF(oEH), WSF(oEC), WSU(oEHB), WSF(oEG), WSU(oEncA), WSU(oEncT),
        WSF(oHfin), WSF(oCfin));

    // 6) attention projections Ea / Ez
    k_gemm<<<ggrid(224*32), 256, 0, stream>>>(WSU(oEncA), pKH, WSU(oWaT), pKH,
        cB*pTE, cH, pKH, (float*)nullptr, 0, WSU(oEA), pKH, (const float*)nullptr);
    k_gemm<<<ggrid(224*32), 256, 0, stream>>>(WSU(oInzA), 64, WSU(oWzT), 64,
        cB*pTE, cH, 64, (float*)nullptr, 0, WSU(oEZ), pKH, (const float*)nullptr);

    // 7) decoder init state (torch .view quirk) + decoder recurrence
    k_dh0<<<64, 256, 0, stream>>>(WSF(oHfin), WSF(oCfin), WSF(oDH), WSF(oDC), WSU(oDHB));
    k_dec_rec<<<1, 512, 0, stream>>>(WSF(oGxD), WSU(oDWhh), WSF(oDH), WSF(oDC),
        WSU(oDHB), WSF(oDG), WSU(oHsA), WSU(oCatA), out);

    // 8) dual attention + context + output projection (one WG per batch)
    k_att<<<cB, 512, 0, stream>>>(WSU(oHsA), WSU(oEA), WSU(oEZ), WSU(oEncT),
        WSU(oAttnA), WSU(oCatA), WSU(oWoT), WSU(oDecA), value, e_tgt, out);

    // 9) fused logits + p_lex over vocab chunks (async LDS staging)
    k_big<<<cB * NCHN, 256, 0, stream>>>(WSU(oDecA), WSU(oLinWT), linb, WSU(oAttnA),
        aprob, xpar, WSF(oLogits), out);

    // 10) log-softmax of logits, then final log-softmax + decoder_pred
    k_lse<<<cB * cTD, 256, 0, stream>>>(WSF(oLogits), WSF(oLogZ));
    k_final<<<cB * cTD, 256, 0, stream>>>(out, WSF(oLogZ), xpar, e_tgt, out);

    #undef WSU
    #undef WSF
}
